// LuongAttention_17179869184254
// MI455X (gfx1250) — matrix operand order repaired and checked by reading, not compiled, not run on hardware
//
#include <hip/hip_runtime.h>
#include <hip/hip_bf16.h>

// ---------------------------------------------------------------------------
// Luong attention for MI455X (gfx1250, wave32, WMMA).
//   z       = enc @ W^T + b            (f16 WMMA, fp32 accum, z stored f16)
//   scores  = dec @ z^T   -> softmax -> weights (fp32 out) -> context
// Fused kernel keeps the 16 x 2048 score stripe entirely in LDS (CDNA5 has
// 320KB/WGP) so scores never touch HBM; only the mandated weights/context
// outputs are written.
// ---------------------------------------------------------------------------

typedef __attribute__((ext_vector_type(16))) _Float16 v16h;
typedef __attribute__((ext_vector_type(8)))  _Float16 v8h;
typedef __attribute__((ext_vector_type(8)))  float    v8f;

static constexpr int Bsz = 32;
static constexpr int Ssz = 2048;
static constexpr int Tsz = 512;
static constexpr int Hsz = 1024;

// LDS row strides (elements), padded so 16-lane fragment accesses hit 16
// distinct banks (64 banks x 4B on CDNA5).
static constexpr int SSP = 2052; // scores rows, fp32
static constexpr int SWP = 2056; // weights rows, f16
static constexpr int DSP = 1032; // decoder/A tile rows, f16
static constexpr int EST = 40;   // transposed E-chunk rows (32 K-vals + pad), f16

static constexpr size_t SMEM_BYTES =
    (size_t)16 * SSP * 4 + (size_t)16 * SWP * 2 +
    (size_t)16 * DSP * 2 + (size_t)Hsz * EST * 2; // = 312064 B <= 320KB

#define HD __device__ __forceinline__

// A-matrix fragment (16x32 f16, MxK): lane r=lane&15 holds row M=r.
// Lanes 0-15 hold K = 0..15, lanes 16-31 hold K = 16..31 (matches B)
// ->  two contiguous 8-half (16B) loads per lane.
HD v16h load_frag_a(const _Float16* base /* row0,k0 */, int ld, int lane) {
    int r = lane & 15, hs = lane >> 4;
    const _Float16* p = base + r * ld + hs * 16;
    v8h lo = *(const v8h*)p;          // K = hs*16 + 0..7
    v8h hi = *(const v8h*)(p + 8);    // K = hs*16 + 8..15
    return __builtin_shufflevector(lo, hi, 0,1,2,3,4,5,6,7,8,9,10,11,12,13,14,15);
}

// B-matrix fragment (32x16 f16, KxN) where B[k][n] = base[n*ld + k]
// (i.e. the stored matrix's row n is B's column n, K contiguous).
// Lanes 0-15: N=lane, K=0..15; lanes 16-31: N=lane-16, K=16..31.
// One contiguous 16-half (32B) load per lane.
HD v16h load_frag_b(const _Float16* base /* n0,k0 */, int ld, int lane) {
    int n = lane & 15, kh = lane >> 4;
    return *(const v16h*)(base + n * ld + kh * 16);
}

// ---------------------------------------------------------------------------
__global__ void cvt_f32_f16(const float* __restrict__ in,
                            _Float16* __restrict__ out, long long n) {
    long long i = (long long)blockIdx.x * blockDim.x + threadIdx.x;
    long long stride = (long long)gridDim.x * blockDim.x;
    for (; i < n; i += stride) out[i] = (_Float16)in[i];
}

// ---------------------------------------------------------------------------
// z = E @ W^T + b  as (B*S x H) x (H x H).
// One block per 16-row m-tile covering all N=1024: the 16x1024 A-tile is
// staged once in LDS and reused by 8 waves x 8 n-tiles (64 WMMA tiles),
// 1 A-frag ds-load amortized over 8 WMMAs.  W (2MB f16) is L2-resident.
__global__ void __launch_bounds__(256)
gemm1_kernel(const _Float16* __restrict__ E, const _Float16* __restrict__ Wh,
             const float* __restrict__ bias, _Float16* __restrict__ Z) {
    __shared__ _Float16 sA[16 * DSP];
    int tid = threadIdx.x, lane = tid & 31, wave = tid >> 5;
    int m0 = blockIdx.x * 16;
    {
        int r = tid >> 4;          // 0..15
        int c = (tid & 15) * 64;   // 0..960
#pragma unroll
        for (int j = 0; j < 64; j += 8)
            *(v8h*)(sA + r * DSP + c + j) =
                *(const v8h*)(E + (size_t)(m0 + r) * Hsz + c + j);
    }
    __syncthreads();

    v8f acc[8];
    {
        v8f z = {};
#pragma unroll
        for (int q = 0; q < 8; ++q) acc[q] = z;
    }
    for (int k0 = 0; k0 < Hsz; k0 += 32) {
        v16h a = load_frag_a(sA + k0, DSP, lane);
#pragma unroll
        for (int q = 0; q < 8; ++q) {
            int n0 = (q * 8 + wave) * 16;
            // B[k][n] = W[n0+n, k]: W rows are B columns, K contiguous.
            v16h bf = load_frag_b(Wh + (size_t)n0 * Hsz + k0, Hsz, lane);
            acc[q] = __builtin_amdgcn_wmma_f32_16x16x32_f16(false, a, false, bf,
                                                            (short)0, acc[q],
                                                            false, false);
        }
    }
    int n = lane & 15, hs = lane >> 4;
#pragma unroll
    for (int q = 0; q < 8; ++q) {
        int n0 = (q * 8 + wave) * 16;
        float bv = bias[n0 + n];
        _Float16* out = Z + (size_t)(m0 + hs * 8) * Hsz + n0 + n;
#pragma unroll
        for (int i = 0; i < 8; ++i) out[(size_t)i * Hsz] = (_Float16)(acc[q][i] + bv);
    }
}

// ---------------------------------------------------------------------------
// Fused: scores -> softmax -> weights(out) + context(out).
// Block = 8 waves = one (batch b, 16 decoder rows) stripe.
__global__ void __launch_bounds__(256)
attn_fused_kernel(const _Float16* __restrict__ Dh, const _Float16* __restrict__ Zh,
                  const _Float16* __restrict__ Eh, float* __restrict__ Wout,
                  float* __restrict__ Cout) {
    extern __shared__ __align__(16) char smem[];
    float*    sS = (float*)smem;                                   // 16 x SSP fp32 scores
    _Float16* sW = (_Float16*)(smem + (size_t)16 * SSP * 4);       // 16 x SWP f16 weights
    _Float16* sD = (_Float16*)((char*)sW + (size_t)16 * SWP * 2);  // 16 x DSP f16 dec tile
    _Float16* sE = (_Float16*)((char*)sD + (size_t)16 * DSP * 2);  // Hsz x EST f16 E^T chunk

    int tid = threadIdx.x, lane = tid & 31, wave = tid >> 5;
    int b = blockIdx.y, t0 = blockIdx.x * 16;
    const _Float16* Db = Dh + ((size_t)b * Tsz + t0) * Hsz;
    const _Float16* Zb = Zh + (size_t)b * Ssz * Hsz;
    const _Float16* Eb = Eh + (size_t)b * Ssz * Hsz;

    // Stage the 16 decoder rows (reused by 128 score tiles) into LDS.
    {
        int r = tid >> 4;          // 0..15
        int c = (tid & 15) * 64;   // 0..960
#pragma unroll
        for (int j = 0; j < 64; j += 8)
            *(v8h*)(sD + r * DSP + c + j) = *(const v8h*)(Db + (size_t)r * Hsz + c + j);
    }
    __syncthreads();

    // ---- Phase 1: scores[t0..t0+15][:] into LDS -------------------------
    // 4 s-tiles per inner group: one A-frag ds-load feeds 4 WMMAs.
#pragma unroll 1
    for (int g = 0; g < 4; ++g) {
        int sbase = (g * 8 + wave) * 64;   // 4 consecutive 16-wide s-tiles
        v8f acc[4];
        {
            v8f z = {};
#pragma unroll
            for (int j = 0; j < 4; ++j) acc[j] = z;
        }
        for (int k0 = 0; k0 < Hsz; k0 += 32) {
            v16h a = load_frag_a(sD + k0, DSP, lane);
#pragma unroll
            for (int j = 0; j < 4; ++j) {
                // B[k][n] = Z[b, s+n, k]: Z rows are B columns, K contiguous.
                v16h bf = load_frag_b(Zb + (size_t)(sbase + j * 16) * Hsz + k0,
                                      Hsz, lane);
                acc[j] = __builtin_amdgcn_wmma_f32_16x16x32_f16(false, a, false, bf,
                                                                (short)0, acc[j],
                                                                false, false);
            }
        }
        int n = lane & 15, hs = lane >> 4;
#pragma unroll
        for (int j = 0; j < 4; ++j)
#pragma unroll
            for (int i = 0; i < 8; ++i)
                sS[(hs * 8 + i) * SSP + sbase + j * 16 + n] = acc[j][i];
    }
    __syncthreads();

    // ---- Phase 2: softmax per row (16-lane half-wave per row) -----------
    {
        int n = lane & 15, hs = lane >> 4;
        int row = wave * 2 + hs;
        float* rp = sS + row * SSP;
        float m = -3.402823466e38f;
        for (int s = n; s < Ssz; s += 16) m = fmaxf(m, rp[s]);
        m = fmaxf(m, __shfl_xor(m, 1));
        m = fmaxf(m, __shfl_xor(m, 2));
        m = fmaxf(m, __shfl_xor(m, 4));
        m = fmaxf(m, __shfl_xor(m, 8));   // xor<16: stays within the half-wave
        float sum = 0.f;
        for (int s = n; s < Ssz; s += 16) {
            float e = __expf(rp[s] - m);
            sum += e;
            rp[s] = e;
        }
        sum += __shfl_xor(sum, 1);
        sum += __shfl_xor(sum, 2);
        sum += __shfl_xor(sum, 4);
        sum += __shfl_xor(sum, 8);
        float inv = 1.0f / sum;
        float* wrow = Wout + ((size_t)b * Tsz + t0 + row) * Ssz;
        for (int s = n; s < Ssz; s += 16) {
            float w = rp[s] * inv;
            sW[row * SWP + s] = (_Float16)w; // f16 copy for the context GEMM
            wrow[s] = w;                     // required fp32 output
        }
    }

    // ---- Phase 3: context = weights @ E, K = S in chunks of 32 ----------
    v8f accs[8];
    {
        v8f z = {};
#pragma unroll
        for (int q = 0; q < 8; ++q) accs[q] = z;
    }
    for (int k0 = 0; k0 < Ssz; k0 += 32) {
        __syncthreads(); // sW ready (1st iter) / sE no longer read (later iters)
        // Stage E[b, k0..k0+31, :] transposed: sE[h][s_local], stride EST.
        {
            int sl = tid >> 3;         // 0..31  (s within chunk)
            int co = (tid & 7) * 128;  // h column group
            if (k0 + 32 < Ssz) {       // prefetch next chunk (global_prefetch_b8)
                __builtin_prefetch(Eb + (size_t)(k0 + 32 + sl) * Hsz + co, 0, 3);
                __builtin_prefetch(Eb + (size_t)(k0 + 32 + sl) * Hsz + co + 64, 0, 3);
            }
#pragma unroll
            for (int c = 0; c < 128; c += 8) {
                v8h v = *(const v8h*)(Eb + (size_t)(k0 + sl) * Hsz + co + c);
#pragma unroll
                for (int i = 0; i < 8; ++i) sE[(co + c + i) * EST + sl] = v[i];
            }
        }
        __syncthreads();
        v16h a = load_frag_a(sW + k0, SWP, lane); // same A for all 8 h-tiles
#pragma unroll
        for (int q = 0; q < 8; ++q) {
            int h0 = (q * 8 + wave) * 16;
            // B[k][n] = E[b, k0+k, h0+n] = sE[(h0+n)*EST + k]
            v16h bf = load_frag_b(sE + (size_t)h0 * EST, EST, lane);
            accs[q] = __builtin_amdgcn_wmma_f32_16x16x32_f16(false, a, false, bf,
                                                             (short)0, accs[q],
                                                             false, false);
        }
    }
    {
        int n = lane & 15, hs = lane >> 4;
#pragma unroll
        for (int q = 0; q < 8; ++q) {
            int h0 = (q * 8 + wave) * 16;
            float* out = Cout + ((size_t)b * Tsz + t0 + hs * 8) * Hsz + h0 + n;
#pragma unroll
            for (int i = 0; i < 8; ++i) out[(size_t)i * Hsz] = accs[q][i];
        }
    }
}

// ---------------------------------------------------------------------------
extern "C" void kernel_launch(void* const* d_in, const int* in_sizes, int n_in,
                              void* d_out, int out_size, void* d_ws, size_t ws_size,
                              hipStream_t stream) {
    const float* enc  = (const float*)d_in[0]; // [B,S,H]
    const float* dec  = (const float*)d_in[1]; // [B,T,H]
    const float* W    = (const float*)d_in[2]; // [H,H]
    const float* bias = (const float*)d_in[3]; // [H]

    const size_t nE = (size_t)Bsz * Ssz * Hsz; // 67.1M
    const size_t nD = (size_t)Bsz * Tsz * Hsz; // 16.8M
    const size_t nW = (size_t)Hsz * Hsz;       // 1.0M

    char* ws = (char*)d_ws;                    // ~304 MB used
    _Float16* Eh = (_Float16*)ws; ws += nE * 2;
    _Float16* Zh = (_Float16*)ws; ws += nE * 2;
    _Float16* Dh = (_Float16*)ws; ws += nD * 2;
    _Float16* Wh = (_Float16*)ws;

    float* Wout = (float*)d_out;                      // [B,T,S] weights
    float* Cout = Wout + (size_t)Bsz * Tsz * Ssz;     // [B,T,H] context

    cvt_f32_f16<<<2048, 256, 0, stream>>>(enc, Eh, (long long)nE);
    cvt_f32_f16<<<512,  256, 0, stream>>>(dec, Dh, (long long)nD);
    cvt_f32_f16<<<64,   256, 0, stream>>>(W,   Wh, (long long)nW);

    // z GEMM: one block per 16-row m-tile, 8 waves x 8 n-tiles cover N=1024.
    gemm1_kernel<<<dim3(Bsz * Ssz / 16), 256, 0, stream>>>(Eh, Wh, bias, Zh);

    (void)hipFuncSetAttribute((const void*)attn_fused_kernel,
                              hipFuncAttributeMaxDynamicSharedMemorySize,
                              (int)SMEM_BYTES);
    attn_fused_kernel<<<dim3(Tsz / 16, Bsz), 256, SMEM_BYTES, stream>>>(
        Dh, Zh, Eh, Wout, Cout);
}
